// S_GRU_38577396253099
// MI455X (gfx1250) — matrix operand order repaired and checked by reading, not compiled, not run on hardware
//
#include <hip/hip_runtime.h>
#include <hip/hip_bf16.h>

#define BB 32
#define TT 512
#define HH 512
#define LL 4
#define BH (BB*HH)          // 16384
#define NBLK 64             // 4 layer-groups * 16 column slices
#define NTHR 256            // 8 waves (wave32)

typedef __attribute__((ext_vector_type(16))) __bf16 v16bf;
typedef __attribute__((ext_vector_type(8)))  float  v8f;

// ---------- helpers ----------
__device__ __forceinline__ unsigned short f2bf(float f) {
  unsigned u = __builtin_bit_cast(unsigned, f);
  u += 0x7FFFu + ((u >> 16) & 1u);          // RNE
  return (unsigned short)(u >> 16);
}
__device__ __forceinline__ float sigmoidf_(float x) { return 1.0f / (1.0f + __expf(-x)); }

union Frag { uint4 u[2]; v16bf v; };

// A-matrix 16x32 bf16 layout: lane(0-15) holds K={kb+0..7, kb+16..23}; lane(16-31): +8
__device__ __forceinline__ v16bf frag_a(const unsigned short* buf, int rowbase, int k, int lane) {
  const unsigned short* p = buf + (rowbase + (lane & 15)) * HH + k + ((lane >> 4) << 3);
  Frag f; f.u[0] = *(const uint4*)p; f.u[1] = *(const uint4*)(p + 16);
  return f.v;
}
// B-matrix 32x16 bf16 (K-major, pre-transposed [n][k]): lane(0-15) holds K={kb+0..7, kb+16..23} of column n; lane(16-31): +8
__device__ __forceinline__ v16bf frag_b(const unsigned short* wcol, int nsub, int k, int lane) {
  const unsigned short* p = wcol + (((nsub << 4) + (lane & 15)) * HH) + k + ((lane >> 4) << 3);
  Frag f; f.u[0] = *(const uint4*)p; f.u[1] = *(const uint4*)(p + 16);
  return f.v;
}
__device__ __forceinline__ v8f wmma_bf16(v16bf a, v16bf b, v8f c) {
  return __builtin_amdgcn_wmma_f32_16x16x32_bf16(false, a, false, b, (short)0, c, false, false);
}

__device__ __forceinline__ void grid_barrier(unsigned* cnt, unsigned* gen) {
  __threadfence();
  __syncthreads();
  if (threadIdx.x == 0) {
    unsigned g = __hip_atomic_load(gen, __ATOMIC_RELAXED, __HIP_MEMORY_SCOPE_AGENT);
    unsigned a = __hip_atomic_fetch_add(cnt, 1u, __ATOMIC_ACQ_REL, __HIP_MEMORY_SCOPE_AGENT);
    if (a == NBLK - 1) {
      __hip_atomic_store(cnt, 0u, __ATOMIC_RELAXED, __HIP_MEMORY_SCOPE_AGENT);
      __hip_atomic_fetch_add(gen, 1u, __ATOMIC_RELEASE, __HIP_MEMORY_SCOPE_AGENT);
    } else {
      while (__hip_atomic_load(gen, __ATOMIC_ACQUIRE, __HIP_MEMORY_SCOPE_AGENT) == g)
        __builtin_amdgcn_s_sleep(1);
    }
  }
  __syncthreads();
}

// ---------- prologue: zero state + barrier ----------
__global__ void zero_ws(unsigned* p, int n) {
  int i = blockIdx.x * NTHR + threadIdx.x;
  if (i < n) p[i] = 0u;
}

// ---------- prologue: transpose+convert weights to bf16 [mat][n][k] ----------
// mat order per layer: U0,U1,U2,W0,W1,W2 ; src U[l,m,k,n], W[l,m,k,n]
__global__ void wt_convert(const float* __restrict__ W, const float* __restrict__ U,
                           unsigned short* __restrict__ wt) {
  int idx = blockIdx.x * NTHR + threadIdx.x;        // (mat, k, n): coalesced reads
  if (idx >= 24 * HH * HH) return;
  int mat = idx >> 18;
  int rem = idx & 262143;
  int k = rem >> 9;
  int n = rem & 511;
  int layer = mat / 6, m6 = mat % 6;
  float v = (m6 < 3) ? U[(((size_t)(layer * 3 + m6) * HH) + k) * HH + n]
                     : W[(((size_t)(layer * 3 + (m6 - 3)) * HH) + k) * HH + n];
  wt[(size_t)mat * (HH * HH) + (size_t)n * HH + k] = f2bf(v);
}

// ---------- persistent wavefront GRU ----------
__global__ void __launch_bounds__(NTHR)
gru_persistent(const float* __restrict__ xin,          // [B,T,D] fp32
               const unsigned short* __restrict__ wt,  // bf16 [L*6][n][k]
               unsigned short* __restrict__ g_sb,      // bf16 state [L][B][H]
               unsigned short* __restrict__ g_rs,      // bf16 r*s    [L][B][H]
               unsigned* __restrict__ bar,
               float* __restrict__ out)                // [B,T,H] fp32
{
  extern __shared__ char smem[];
  unsigned short* sm_w  = (unsigned short*)smem;     // 6*32*512 bf16 = 192KB weight slice
  unsigned short* sm_x  = sm_w + 6 * 32 * HH;        // layer input  [32][512] bf16
  unsigned short* sm_p  = sm_x + BH;                 // prev state   [32][512] bf16
  unsigned short* sm_rs = sm_p + BH;                 // r*s staged   [32][512] bf16
  float* sm_z  = (float*)(sm_rs + BH);               // z  [32][32]
  float* sm_xh = sm_z + 1024;                        // xh [32][32]
  float* sm_sp = sm_xh + 1024;                       // persistent fp32 state, own cols [32][32]

  const int tid  = threadIdx.x;
  const int wid  = tid >> 5, lane = tid & 31;
  const int grp  = blockIdx.x >> 4;                  // layer handled by this WG
  const int c0   = (blockIdx.x & 15) << 5;           // owned global column base (32 cols)

  // one-time: weight slice -> LDS (192KB = 12288 uint4)
  for (int i = tid; i < 12288; i += NTHR) {
    int matcol = i >> 6, off = i & 63;
    int mat = matcol >> 5, col = matcol & 31;
    const unsigned short* s =
        wt + ((size_t)((grp * 6 + mat) * HH + c0 + col)) * HH + (off << 3);
    ((uint4*)sm_w)[i] = *(const uint4*)s;
  }
  for (int i = tid; i < 1024; i += NTHR) sm_sp[i] = 0.0f;   // s(t=-1) = 0
  __syncthreads();

  for (int step = 0; step < TT + LL - 1; ++step) {
    const int t = step - grp;
    const bool active = (t >= 0) && (t < TT);

    if (active) {
      // stage x (layer input) and prev-state bf16 into LDS
      if (grp == 0) {
        for (int i = tid; i < 4096; i += NTHR) {           // float4 units
          int b = i >> 7, k4 = (i & 127) << 2;
          const float4 f = *(const float4*)(xin + ((size_t)b * TT + t) * HH + k4);
          unsigned short* d = sm_x + b * HH + k4;
          d[0] = f2bf(f.x); d[1] = f2bf(f.y); d[2] = f2bf(f.z); d[3] = f2bf(f.w);
        }
      } else {
        const uint4* s4 = (const uint4*)(g_sb + (size_t)(grp - 1) * BH);
        for (int i = tid; i < BH / 8; i += NTHR) ((uint4*)sm_x)[i] = s4[i];
      }
      {
        const uint4* s4 = (const uint4*)(g_sb + (size_t)grp * BH);
        for (int i = tid; i < BH / 8; i += NTHR) ((uint4*)sm_p)[i] = s4[i];
      }
      __syncthreads();

      // ---- Phase A: z, r, xh for owned 32 columns ----
      if (wid < 4) {
        // waves 0..3: gate z (g=0) or r (g=1), two row tiles each; 4 WMMA / k-step,
        // straight-line + software pipelined (prefetch k+32, wrap on last iter)
        const int g = wid >> 1, m = wid & 1;
        const unsigned short* wu = sm_w + (size_t)g * (32 * HH);        // U_g
        const unsigned short* ww = sm_w + (size_t)(3 + g) * (32 * HH);  // W_g
        v8f a0 = {}, a1 = {};
        v16bf ax = frag_a(sm_x, m * 16, 0, lane);
        v16bf ap = frag_a(sm_p, m * 16, 0, lane);
        v16bf b0 = frag_b(wu, 0, 0, lane);
        v16bf b1 = frag_b(wu, 1, 0, lane);
        v16bf b2 = frag_b(ww, 0, 0, lane);
        v16bf b3 = frag_b(ww, 1, 0, lane);
        #pragma unroll 4
        for (int k = 0; k < HH; k += 32) {
          const int kn = (k + 32) & (HH - 1);     // last prefetch wraps (discarded)
          v16bf ax_n = frag_a(sm_x, m * 16, kn, lane);
          v16bf ap_n = frag_a(sm_p, m * 16, kn, lane);
          v16bf b0_n = frag_b(wu, 0, kn, lane);
          v16bf b1_n = frag_b(wu, 1, kn, lane);
          v16bf b2_n = frag_b(ww, 0, kn, lane);
          v16bf b3_n = frag_b(ww, 1, kn, lane);
          a0 = wmma_bf16(ax, b0, a0);
          a1 = wmma_bf16(ax, b1, a1);
          a0 = wmma_bf16(ap, b2, a0);
          a1 = wmma_bf16(ap, b3, a1);
          ax = ax_n; ap = ap_n; b0 = b0_n; b1 = b1_n; b2 = b2_n; b3 = b3_n;
        }
        const int rbase = m * 16 + ((lane >> 4) << 3);
        const int cl = lane & 15;
        #pragma unroll
        for (int v = 0; v < 8; ++v) {
          int r = rbase + v;
          if (g == 0) {
            sm_z[r * 32 + cl]      = sigmoidf_(a0[v]);
            sm_z[r * 32 + 16 + cl] = sigmoidf_(a1[v]);
          } else {
            float v0 = sigmoidf_(a0[v]) * sm_sp[r * 32 + cl];
            float v1 = sigmoidf_(a1[v]) * sm_sp[r * 32 + 16 + cl];
            g_rs[(size_t)grp * BH + r * HH + c0 + cl]      = f2bf(v0);
            g_rs[(size_t)grp * BH + r * HH + c0 + 16 + cl] = f2bf(v1);
          }
        }
      } else if (wid < 6) {
        // waves 4,5: xh = x @ U2 ; 2 WMMA / k-step, pipelined
        const int m = wid & 1;
        const unsigned short* wu = sm_w + (size_t)2 * (32 * HH);        // U_2
        v8f a0 = {}, a1 = {};
        v16bf ax = frag_a(sm_x, m * 16, 0, lane);
        v16bf b0 = frag_b(wu, 0, 0, lane);
        v16bf b1 = frag_b(wu, 1, 0, lane);
        #pragma unroll 4
        for (int k = 0; k < HH; k += 32) {
          const int kn = (k + 32) & (HH - 1);
          v16bf ax_n = frag_a(sm_x, m * 16, kn, lane);
          v16bf b0_n = frag_b(wu, 0, kn, lane);
          v16bf b1_n = frag_b(wu, 1, kn, lane);
          a0 = wmma_bf16(ax, b0, a0);
          a1 = wmma_bf16(ax, b1, a1);
          ax = ax_n; b0 = b0_n; b1 = b1_n;
        }
        const int rbase = m * 16 + ((lane >> 4) << 3);
        const int cl = lane & 15;
        #pragma unroll
        for (int v = 0; v < 8; ++v) {
          int r = rbase + v;
          sm_xh[r * 32 + cl]      = a0[v];
          sm_xh[r * 32 + 16 + cl] = a1[v];
        }
      }
    }
    grid_barrier(bar, bar + 64);

    if (active) {
      // stage full r*s vector for this layer
      const uint4* s4 = (const uint4*)(g_rs + (size_t)grp * BH);
      for (int i = tid; i < BH / 8; i += NTHR) ((uint4*)sm_rs)[i] = s4[i];
      __syncthreads();

      // ---- Phase B: h = tanh(xh + rs@W2), s = (1-z)h + z*s_prev ----
      if (wid < 4) {
        const int m = wid & 1, ns = wid >> 1;
        const unsigned short* w2 = sm_w + (size_t)5 * (32 * HH);
        const int rbase = m * 16 + ((lane >> 4) << 3);
        const int cl = (ns << 4) + (lane & 15);
        v8f acc;
        #pragma unroll
        for (int v = 0; v < 8; ++v) acc[v] = sm_xh[(rbase + v) * 32 + cl];
        v16bf ar = frag_a(sm_rs, m * 16, 0, lane);
        v16bf bw = frag_b(w2, ns, 0, lane);
        #pragma unroll 4
        for (int k = 0; k < HH; k += 32) {
          const int kn = (k + 32) & (HH - 1);
          v16bf ar_n = frag_a(sm_rs, m * 16, kn, lane);
          v16bf bw_n = frag_b(w2, ns, kn, lane);
          acc = wmma_bf16(ar, bw, acc);
          ar = ar_n; bw = bw_n;
        }
        #pragma unroll
        for (int v = 0; v < 8; ++v) {
          int r = rbase + v;
          float z  = sm_z[r * 32 + cl];
          float sp = sm_sp[r * 32 + cl];
          float h  = tanhf(acc[v]);
          float s  = (1.0f - z) * h + z * sp;
          sm_sp[r * 32 + cl] = s;                                  // fp32 state lives in LDS
          g_sb[(size_t)grp * BH + r * HH + c0 + cl] = f2bf(s);     // bf16 state for GEMMs
          if (grp == LL - 1) out[((size_t)r * TT + t) * HH + c0 + cl] = s;
        }
      }
    }
    grid_barrier(bar, bar + 64);
  }
}

// ---------- host ----------
extern "C" void kernel_launch(void* const* d_in, const int* in_sizes, int n_in,
                              void* d_out, int out_size, void* d_ws, size_t ws_size,
                              hipStream_t stream) {
  const float* in = (const float*)d_in[0];   // [32,512,512]
  const float* W  = (const float*)d_in[1];   // [4,3,512,512]
  const float* U  = (const float*)d_in[2];   // [4,3,512,512]
  float* out = (float*)d_out;
  char* ws = (char*)d_ws;

  unsigned short* wt   = (unsigned short*)ws;                        // 12,582,912 B
  unsigned short* g_sb = (unsigned short*)(ws + 12582912);           // 131,072 B
  unsigned short* g_rs = (unsigned short*)(ws + 12582912 + 131072);  // 131,072 B
  unsigned* bar = (unsigned*)(ws + 12582912 + 131072 + 131072);      // 512 B (cnt @ +0, gen @ +256)

  // zero state buffers + barrier (run every launch -> deterministic)
  int nz = (131072 + 131072 + 512) / 4;
  zero_ws<<<(nz + NTHR - 1) / NTHR, NTHR, 0, stream>>>((unsigned*)(ws + 12582912), nz);

  // weights -> bf16, transposed to [n][k]
  wt_convert<<<(24 * HH * HH) / NTHR, NTHR, 0, stream>>>(W, U, wt);

  const int smem = 307200;  // 300KB: 192KB weights + 3x32KB act + 3x4KB tiles
  hipFuncSetAttribute((const void*)gru_persistent,
                      hipFuncAttributeMaxDynamicSharedMemorySize, smem);
  gru_persistent<<<NBLK, NTHR, smem, stream>>>(in, wt, g_sb, g_rs, bar, out);
}
